// MultiheadSelfAttention_16346645528922
// MI455X (gfx1250) — matrix operand order repaired and checked by reading, not compiled, not run on hardware
//
#include <hip/hip_runtime.h>

// ---------------------------------------------------------------------------
// MHA with RoPE for MI455X (gfx1250, wave32, WMMA bf16->f32)
// B=2, S=2048, D=768, H=12, dk=64
// ---------------------------------------------------------------------------

#define S_LEN   2048
#define B_SZ    2
#define D_MOD   768
#define NHEAD   12
#define DK      64
#define MROWS   (B_SZ * S_LEN)     // 4096
#define NQKV    (3 * D_MOD)        // 2304

typedef __attribute__((ext_vector_type(8)))  __bf16 v8bf;
typedef __attribute__((ext_vector_type(16))) __bf16 v16bf;
typedef __attribute__((ext_vector_type(8)))  float  v8f;
typedef __attribute__((ext_vector_type(4)))  int    v4i;

// ---- CDNA5 async global->LDS path (guarded; sync fallback keeps compile safe)
#if defined(__has_builtin)
#if __has_builtin(__builtin_amdgcn_global_load_async_to_lds_b128) && \
    __has_builtin(__builtin_amdgcn_s_wait_asynccnt)
#define USE_ASYNC_LDS 1
#endif
#endif
#ifndef USE_ASYNC_LDS
#define USE_ASYNC_LDS 0
#endif

#if USE_ASYNC_LDS
typedef __attribute__((address_space(1))) v4i gas_v4i;
typedef __attribute__((address_space(3))) v4i las_v4i;
#define WAIT_ASYNC(n) __builtin_amdgcn_s_wait_asynccnt(n)
#else
#define WAIT_ASYNC(n)
#endif

// copy one 16B chunk global -> LDS
__device__ __forceinline__ void stage_chunk(const __bf16* src, __bf16* dst) {
#if USE_ASYNC_LDS
  __builtin_amdgcn_global_load_async_to_lds_b128((gas_v4i*)src, (las_v4i*)dst,
                                                 0, 0);
#else
  *(v8bf*)dst = *(const v8bf*)src;
#endif
}

__device__ __forceinline__ v8f wmma_bf16(v16bf a, v16bf b, v8f c) {
  return __builtin_amdgcn_wmma_f32_16x16x32_bf16(false, a, false, b, (short)0, c,
                                                 false, false);
}

// Assemble a 16-element bf16 fragment from two 8-element chunks p[0..7], p[gap..gap+7]
__device__ __forceinline__ v16bf frag_gap(const __bf16* p, int gap) {
  v8bf lo = *(const v8bf*)p;
  v8bf hi = *(const v8bf*)(p + gap);
  v16bf r;
#pragma unroll
  for (int i = 0; i < 8; ++i) { r[i] = lo[i]; r[i + 8] = hi[i]; }
  return r;
}

// ---------------------------------------------------------------------------
// f32 -> bf16 convert
// ---------------------------------------------------------------------------
__global__ void f32_to_bf16_kernel(const float* __restrict__ in,
                                   __bf16* __restrict__ out, int n) {
  int i = blockIdx.x * blockDim.x + threadIdx.x;
  if (i < n) out[i] = (__bf16)in[i];
}

// ---------------------------------------------------------------------------
// GEMM: C[M,N] = A[M,K] * W[N,K]^T   (bf16 in, f32 out)
// Block 256 thr (8 waves), tile 128x256x32, double-buffered async LDS staging.
// Wave = 64x64 (4x4 WMMA tiles): 16 WMMAs per 8 fragment loads per K-step.
// ---------------------------------------------------------------------------
#define LDT 40  // padded LDS row stride in elements (80B, still 16B aligned)

__global__ __launch_bounds__(256) void gemm_bf16_tn(
    const __bf16* __restrict__ A, const __bf16* __restrict__ W,
    float* __restrict__ C, int M, int N, int K) {
  __shared__ __bf16 Al[2][128 * LDT];  // 2 x 10 KB
  __shared__ __bf16 Bl[2][256 * LDT];  // 2 x 20 KB

  const int tid  = threadIdx.x;
  const int lane = tid & 31;
  const int wid  = tid >> 5;
  const int half = lane >> 4;
  const int ln   = lane & 15;
  const int bm   = blockIdx.y * 128;
  const int bn   = blockIdx.x * 256;
  const int mw   = (wid >> 2) * 64;  // wave row offset in tile
  const int nw   = (wid & 3) * 64;   // wave col offset in tile

  v8f acc[4][4];
#pragma unroll
  for (int s = 0; s < 4; ++s)
#pragma unroll
    for (int j = 0; j < 4; ++j) acc[s][j] = (v8f)0.0f;

  // Stage one 128x32 A tile + 256x32 W tile: 1536 16B chunks, 6 per thread.
  auto issue_tile = [&](int k0, int buf) {
#pragma unroll
    for (int c = 0; c < 2; ++c) {  // A: chunks 0..511
      int ch  = tid + c * 256;
      int row = ch >> 2;
      int col = (ch & 3) * 8;
      stage_chunk(&A[(size_t)(bm + row) * K + k0 + col],
                  &Al[buf][row * LDT + col]);
    }
#pragma unroll
    for (int c = 0; c < 4; ++c) {  // B: chunks 0..1023
      int ch  = tid + c * 256;
      int row = ch >> 2;
      int col = (ch & 3) * 8;
      stage_chunk(&W[(size_t)(bn + row) * K + k0 + col],
                  &Bl[buf][row * LDT + col]);
    }
  };

  const int nk = K / 32;
  issue_tile(0, 0);

  for (int kt = 0; kt < nk; ++kt) {
    const int buf = kt & 1;
    if (kt + 1 < nk) {
      issue_tile((kt + 1) * 32, buf ^ 1);
      WAIT_ASYNC(6);  // in-order: previous tile's 6 ops are complete
    } else {
      WAIT_ASYNC(0);
    }
    __syncthreads();

    v16bf af[4], bfr[4];
#pragma unroll
    for (int s = 0; s < 4; ++s)
      af[s] = frag_gap(&Al[buf][(mw + s * 16 + ln) * LDT + half * 8], 16);
#pragma unroll
    for (int j = 0; j < 4; ++j)
      bfr[j] = frag_gap(&Bl[buf][(nw + j * 16 + ln) * LDT + half * 8], 16);

#pragma unroll
    for (int s = 0; s < 4; ++s)
#pragma unroll
      for (int j = 0; j < 4; ++j)
        acc[s][j] = wmma_bf16(af[s], bfr[j], acc[s][j]);
    __syncthreads();  // protect buffer before next overwrite
  }

  // Epilogue: C layout — VGPR r: row = r + 8*half, col = lane&15.
#pragma unroll
  for (int s = 0; s < 4; ++s)
#pragma unroll
    for (int j = 0; j < 4; ++j)
#pragma unroll
      for (int r = 0; r < 8; ++r) {
        int row = bm + mw + s * 16 + r + 8 * half;
        int col = bn + nw + j * 16 + ln;
        C[(size_t)row * N + col] = acc[s][j][r];
      }
}

// ---------------------------------------------------------------------------
// RoPE + repack: QKV f32 [4096,2304] -> Qh/Kh bf16 [B,H,S,dk], Vt bf16 [B,H,dk,S]
// One thread per rotation pair (i in 0..31) per head per row.
// ---------------------------------------------------------------------------
__global__ void rope_pack_kernel(const float* __restrict__ QKV,
                                 __bf16* __restrict__ Qh,
                                 __bf16* __restrict__ Kh,
                                 __bf16* __restrict__ Vt) {
  int t = blockIdx.x * blockDim.x + threadIdx.x;
  if (t >= MROWS * NHEAD * 32) return;
  int i   = t & 31;
  int h   = (t >> 5) % NHEAD;
  int row = t / (NHEAD * 32);
  int b   = row >> 11;        // /2048
  int s   = row & 2047;

  const float* p = QKV + (size_t)row * NQKV + h * DK + 2 * i;
  float q1 = p[0],         q2 = p[1];
  float k1 = p[D_MOD],     k2 = p[D_MOD + 1];
  float v1 = p[2 * D_MOD], v2 = p[2 * D_MOD + 1];

  float freq = __powf(10000.0f, -(float)(2 * i) / (float)DK);
  float ang  = (float)s * freq;
  float sn, cs;
  __sincosf(ang, &sn, &cs);

  size_t bh = (size_t)(b * NHEAD + h);
  size_t qo = (bh * S_LEN + s) * DK + 2 * i;
  Qh[qo]     = (__bf16)(q1 * cs - q2 * sn);
  Qh[qo + 1] = (__bf16)(q1 * sn + q2 * cs);
  Kh[qo]     = (__bf16)(k1 * cs - k2 * sn);
  Kh[qo + 1] = (__bf16)(k1 * sn + k2 * cs);

  size_t vo = (bh * DK + 2 * i) * S_LEN + s;
  Vt[vo]         = (__bf16)v1;
  Vt[vo + S_LEN] = (__bf16)v2;
}

// ---------------------------------------------------------------------------
// Flash attention (causal). Grid (S/64, B*H); block 128 thr = 4 waves.
// Wave w owns 16 query rows. Key tiles of 64, online softmax.
// ---------------------------------------------------------------------------
__global__ __launch_bounds__(128) void attn_kernel(
    const __bf16* __restrict__ Qh, const __bf16* __restrict__ Kh,
    const __bf16* __restrict__ Vt, __bf16* __restrict__ Oattn) {
  __shared__ __bf16 Pl[4][16 * 64];

  const int lane = threadIdx.x & 31;
  const int w    = threadIdx.x >> 5;
  const int half = lane >> 4;
  const int ln   = lane & 15;
  const int bh   = blockIdx.y;
  const int b    = bh / NHEAD;
  const int h    = bh % NHEAD;
  const int qt   = blockIdx.x;
  const int q0   = qt * 64 + w * 16;

  const __bf16* Qb = Qh + (size_t)bh * S_LEN * DK;
  const __bf16* Kb = Kh + (size_t)bh * S_LEN * DK;
  const __bf16* Vb = Vt + (size_t)bh * DK * S_LEN;

  // Q A-fragments (16x32 each, dk chunks 0..31 / 32..63)
  const __bf16* qp = Qb + (size_t)(q0 + ln) * DK;
  v16bf qf0 = frag_gap(qp + half * 8, 16);
  v16bf qf1 = frag_gap(qp + 32 + half * 8, 16);

  float mrow[8], lrow[8];
  v8f Oacc[4];
#pragma unroll
  for (int r = 0; r < 8; ++r) { mrow[r] = -__builtin_inff(); lrow[r] = 0.0f; }
#pragma unroll
  for (int jd = 0; jd < 4; ++jd) Oacc[jd] = (v8f)0.0f;

  for (int kt = 0; kt <= qt; ++kt) {
    const int kc0   = kt * 64;
    const bool diag = (kt == qt);

    // ---- scores: S = Q * K^T, 4 column tiles of 16 keys
    v8f Sf[4];
#pragma unroll
    for (int j = 0; j < 4; ++j) {
      const __bf16* kp = Kb + (size_t)(kc0 + j * 16 + ln) * DK + half * 8;
      v16bf kf0 = frag_gap(kp, 16);
      v16bf kf1 = frag_gap(kp + 32, 16);
      v8f s = wmma_bf16(qf0, kf0, (v8f)0.0f);
      Sf[j]  = wmma_bf16(qf1, kf1, s);
    }

    // ---- scale + causal mask
#pragma unroll
    for (int j = 0; j < 4; ++j)
#pragma unroll
      for (int r = 0; r < 8; ++r) {
        float v = Sf[j][r] * 0.125f;  // 1/sqrt(64)
        if (diag && (kc0 + j * 16 + ln) > (q0 + r + 8 * half)) v = -1.0e30f;
        Sf[j][r] = v;
      }

    // ---- online softmax (row reductions across the 16-lane half)
#pragma unroll
    for (int r = 0; r < 8; ++r) {
      float mx = fmaxf(fmaxf(Sf[0][r], Sf[1][r]), fmaxf(Sf[2][r], Sf[3][r]));
      mx = fmaxf(mx, __shfl_xor(mx, 1, 32));
      mx = fmaxf(mx, __shfl_xor(mx, 2, 32));
      mx = fmaxf(mx, __shfl_xor(mx, 4, 32));
      mx = fmaxf(mx, __shfl_xor(mx, 8, 32));
      float mnew = fmaxf(mrow[r], mx);
      float corr = __expf(mrow[r] - mnew);
      float rs = 0.0f;
#pragma unroll
      for (int j = 0; j < 4; ++j) {
        float pv = __expf(Sf[j][r] - mnew);
        Sf[j][r] = pv;
        rs += pv;
      }
      rs += __shfl_xor(rs, 1, 32);
      rs += __shfl_xor(rs, 2, 32);
      rs += __shfl_xor(rs, 4, 32);
      rs += __shfl_xor(rs, 8, 32);
      lrow[r] = lrow[r] * corr + rs;
      mrow[r] = mnew;
#pragma unroll
      for (int jd = 0; jd < 4; ++jd) Oacc[jd][r] *= corr;
    }

    // ---- P: C-layout -> LDS -> A-fragment layout (same-wave DS, in order)
    __bf16* Pw = &Pl[w][0];
#pragma unroll
    for (int j = 0; j < 4; ++j)
#pragma unroll
      for (int r = 0; r < 8; ++r)
        Pw[(r + 8 * half) * 64 + j * 16 + ln] = (__bf16)Sf[j][r];

    const __bf16* Pr = Pw + ln * 64;
    v16bf pf0 = frag_gap(Pr + half * 8, 16);
    v16bf pf1 = frag_gap(Pr + 32 + half * 8, 16);

    // ---- O += P * V (V transposed: [dk, S], contiguous keys per lane)
#pragma unroll
    for (int jd = 0; jd < 4; ++jd) {
      const __bf16* vp = Vb + (size_t)(jd * 16 + ln) * S_LEN + kc0 + half * 8;
      v16bf vf0 = frag_gap(vp, 16);
      v16bf vf1 = frag_gap(vp + 32, 16);
      Oacc[jd] = wmma_bf16(pf0, vf0, Oacc[jd]);
      Oacc[jd] = wmma_bf16(pf1, vf1, Oacc[jd]);
    }
  }

  // ---- normalize + store to [B,S,D] bf16 (col = h*64 + dim)
#pragma unroll
  for (int jd = 0; jd < 4; ++jd)
#pragma unroll
    for (int r = 0; r < 8; ++r) {
      float val = Oacc[jd][r] / lrow[r];
      int qrow  = q0 + r + 8 * half;
      int col   = h * DK + jd * 16 + ln;
      Oattn[(size_t)(b * S_LEN + qrow) * D_MOD + col] = (__bf16)val;
    }
}

// ---------------------------------------------------------------------------
// Host launcher
// ---------------------------------------------------------------------------
extern "C" void kernel_launch(void* const* d_in, const int* in_sizes, int n_in,
                              void* d_out, int out_size, void* d_ws,
                              size_t ws_size, hipStream_t stream) {
  const float* x  = (const float*)d_in[0];
  const float* Wq = (const float*)d_in[1];
  const float* Wk = (const float*)d_in[2];
  const float* Wv = (const float*)d_in[3];
  const float* Wo = (const float*)d_in[4];
  float* out = (float*)d_out;

  // Workspace layout (all sizes already 256B aligned)
  char* ws = (char*)d_ws;
  size_t off = 0;
  __bf16* Xb    = (__bf16*)(ws + off); off += (size_t)MROWS * D_MOD * 2;   // 6.3 MB
  __bf16* Wqkv  = (__bf16*)(ws + off); off += (size_t)NQKV * D_MOD * 2;    // 3.5 MB
  __bf16* Wob   = (__bf16*)(ws + off); off += (size_t)D_MOD * D_MOD * 2;   // 1.2 MB
  float*  QKV   = (float*) (ws + off); off += (size_t)MROWS * NQKV * 4;    // 37.7 MB
  __bf16* Qh    = (__bf16*)(ws + off); off += (size_t)MROWS * D_MOD * 2;
  __bf16* Kh    = (__bf16*)(ws + off); off += (size_t)MROWS * D_MOD * 2;
  __bf16* Vt    = (__bf16*)(ws + off); off += (size_t)MROWS * D_MOD * 2;
  __bf16* Oattn = (__bf16*)(ws + off); off += (size_t)MROWS * D_MOD * 2;

  const int nX = MROWS * D_MOD;      // 3145728
  const int nW = D_MOD * D_MOD;      // 589824
  f32_to_bf16_kernel<<<(nX + 255) / 256, 256, 0, stream>>>(x, Xb, nX);
  f32_to_bf16_kernel<<<(nW + 255) / 256, 256, 0, stream>>>(Wq, Wqkv, nW);
  f32_to_bf16_kernel<<<(nW + 255) / 256, 256, 0, stream>>>(Wk, Wqkv + nW, nW);
  f32_to_bf16_kernel<<<(nW + 255) / 256, 256, 0, stream>>>(Wv, Wqkv + 2 * nW, nW);
  f32_to_bf16_kernel<<<(nW + 255) / 256, 256, 0, stream>>>(Wo, Wob, nW);

  // Fused QKV projection: [4096,768] x [2304,768]^T -> [4096,2304] f32
  gemm_bf16_tn<<<dim3(NQKV / 256, MROWS / 128), 256, 0, stream>>>(
      Xb, Wqkv, QKV, MROWS, NQKV, D_MOD);

  // RoPE + head-major repack (+ V transpose)
  const int nPairs = MROWS * NHEAD * 32;
  rope_pack_kernel<<<(nPairs + 255) / 256, 256, 0, stream>>>(QKV, Qh, Kh, Vt);

  // Causal flash attention
  attn_kernel<<<dim3(S_LEN / 64, B_SZ * NHEAD), 128, 0, stream>>>(Qh, Kh, Vt,
                                                                  Oattn);

  // Output projection: [4096,768] x [768,768]^T -> d_out f32
  gemm_bf16_tn<<<dim3(D_MOD / 256, MROWS / 128), 256, 0, stream>>>(
      Oattn, Wob, out, MROWS, D_MOD, D_MOD);
}
